// SocialImplicit_20298015441094
// MI455X (gfx1250) — matrix operand order repaired and checked by reading, not compiled, not run on hardware
//
#include <hip/hip_runtime.h>

// ---------------------------------------------------------------------------
// SocialImplicit on gfx1250 (MI455X, wave32, WMMA).
//
// Pipeline:
//   k_setup  (1 block): bucketize peds into 4 bins, stable-compact into a
//            guard-padded slot array, build fused 24x64 stage-2 weights (f16)
//            and biases per bin.
//   k_stage1 (VALU): per (k, slot) compute global stage-1 (relu 3x3 + 1x1)
//            and local stage-1 (relu conv1d + 1x1) -> 32 f16 features/slot.
//            Guard slots get zeros (== conv zero padding at bin edges).
//   k_stage2 (WMMA): per (k, 16-position tile) one 24x64 x 64x16 GEMM via
//            4x v_wmma_f32_16x16x32_f16; scatter + bias to d_out.
// ---------------------------------------------------------------------------

#define NPED  50000
#define SLOTS (NPED + 9)      // 1 guard + bin0 + 2 guards + bin1 + ... + tail
#define KST   20

// d_ws layout (bytes)
#define HDR_OFF  0            // int hdr[32]: [0..3]=M, [4..7]=boff, [8..11]=tileStart, [13]=totalTiles
#define BIAS_OFF 256          // float bias2[4][32]
#define W2F_OFF  768          // _Float16 W2f[4][32][64]  (16384 B)
#define IDX_OFF  17408        // int idx_slot[SLOTS]      (200036 B)
#define X1_OFF   217600       // _Float16 x1[kpass][SLOTS][32]  (64B/slot, 64B aligned)

typedef _Float16 v16h __attribute__((ext_vector_type(16)));
typedef _Float16 v8h  __attribute__((ext_vector_type(8)));
typedef float    v8f  __attribute__((ext_vector_type(8)));

__device__ __forceinline__ unsigned ballot32(bool p) {
#if defined(__has_builtin)
#if __has_builtin(__builtin_amdgcn_ballot_w32)
  return __builtin_amdgcn_ballot_w32(p);
#else
  return (unsigned)__ballot(p);
#endif
#else
  return (unsigned)__ballot(p);
#endif
}

__device__ __forceinline__ int bucket_of(const float* v, int p) {
  float nm = fmaxf(fabsf(v[p]), fabsf(v[8 * NPED + p]));
  return (nm >= 0.01f) + (nm >= 0.1f) + (nm >= 1.2f);
}

// ---------------------------------------------------------------------------
// Kernel 1: bucketize + stable compaction + fused stage-2 weight build
// ---------------------------------------------------------------------------
__global__ __launch_bounds__(1024) void k_setup(
    const float* __restrict__ v,
    const float* __restrict__ gtw, const float* __restrict__ gtb,
    const float* __restrict__ ghw, const float* __restrict__ ghb,
    const float* __restrict__ ltw, const float* __restrict__ ltb,
    const float* __restrict__ lhw, const float* __restrict__ lhb,
    const float* __restrict__ gw,  const float* __restrict__ lw,
    char* __restrict__ ws)
{
  int* hdr      = (int*)(ws + HDR_OFF);
  float* bias2  = (float*)(ws + BIAS_OFF);
  _Float16* w2f = (_Float16*)(ws + W2F_OFF);
  int* idx      = (int*)(ws + IDX_OFF);

  __shared__ int cnt[4];
  __shared__ int wcnt[32][4];
  __shared__ int wpre[32][4];
  __shared__ int tot[4];
  __shared__ int base[4];
  __shared__ int sBoff[4];

  const int tid = threadIdx.x;
  if (tid < 4) { cnt[tid] = 0; base[tid] = 0; }
  __syncthreads();

  // ---- histogram ----
  for (int p = tid; p < NPED; p += 1024)
    atomicAdd(&cnt[bucket_of(v, p)], 1);
  __syncthreads();

  if (tid == 0) {
    int boff = 1, ts = 0;
    for (int b = 0; b < 4; b++) {
      int M = cnt[b];
      hdr[b] = M;
      hdr[4 + b] = boff; sBoff[b] = boff;
      hdr[8 + b] = ts;
      boff += M + 2;
      ts += (M + 15) / 16;
    }
    hdr[12] = ts;
    hdr[13] = ts;  // total tiles
  }
  // ---- init all slots to guard (-1) ----
  for (int s = tid; s < SLOTS; s += 1024) idx[s] = -1;
  __syncthreads();

  // ---- stable compaction: chunked block-wide 4-way prefix ----
  const int lane = tid & 31, wv = tid >> 5;
  const unsigned lmask = (1u << lane) - 1u;
  const int nchunks = (NPED + 1023) / 1024;
  for (int ch = 0; ch < nchunks; ch++) {
    int p = ch * 1024 + tid;
    int b = (p < NPED) ? bucket_of(v, p) : 4;
    unsigned m0 = ballot32(b == 0), m1 = ballot32(b == 1),
             m2 = ballot32(b == 2), m3 = ballot32(b == 3);
    if (lane < 4) {
      unsigned mm = (lane == 0) ? m0 : (lane == 1) ? m1 : (lane == 2) ? m2 : m3;
      wcnt[wv][lane] = __builtin_popcount(mm);
    }
    __syncthreads();
    if (tid < 4) {
      int run = 0;
      for (int i = 0; i < 32; i++) { wpre[i][tid] = run; run += wcnt[i][tid]; }
      tot[tid] = run;
    }
    __syncthreads();
    if (b < 4) {
      unsigned mym = (b == 0) ? m0 : (b == 1) ? m1 : (b == 2) ? m2 : m3;
      int pos = base[b] + wpre[wv][b] + __builtin_popcount(mym & lmask);
      idx[sBoff[b] + pos] = p;
    }
    __syncthreads();
    if (tid < 4) base[tid] += tot[tid];
    __syncthreads();
  }

  // ---- fused stage-2 weights: W2f[b][m=h*12+o][f], f = du*16+c*8+t | 48+c*8+t
  for (int e = tid; e < 4 * 32 * 64; e += 1024) {
    int b = e >> 11, m = (e >> 6) & 31, f = e & 63;
    float wvl = 0.f;
    if (m < 24) {
      int h = m / 12, o = m % 12;
      if (f < 48) {
        int du = f >> 4, r = f & 15, c = r >> 3, t = r & 7;
        int kh = c - h + 1;  // only valid taps of the 3x3 H-conv
        float g = gtw[(((b * 12 + o) * 8 + t) * 3 + kh) * 3 + du];
        if (du == 1 && c == h) g += ghw[(b * 12 + o) * 8 + t];
        wvl = gw[b] * g;
      } else {
        int r = f - 48, c = r >> 3, t = r & 7;
        int kh = c - h + 1;
        float g = ltw[((b * 12 + o) * 8 + t) * 3 + kh];
        if (c == h) g += lhw[(b * 12 + o) * 8 + t];
        wvl = lw[b] * g;
      }
    }
    w2f[e] = (_Float16)wvl;
  }
  for (int e = tid; e < 128; e += 1024) {
    int b = e >> 5, m = e & 31;
    float bv = 0.f;
    if (m < 24) {
      int o = m % 12;
      bv = gw[b] * (gtb[b * 12 + o] + ghb[b * 12 + o]) +
           lw[b] * (ltb[b * 12 + o] + lhb[b * 12 + o]);
    }
    bias2[e] = bv;
  }
}

// ---------------------------------------------------------------------------
// Kernel 2: stage-1 features (f16) per (k, slot)
// ---------------------------------------------------------------------------
__global__ __launch_bounds__(256) void k_stage1(
    const float* __restrict__ v, const float* __restrict__ noise,
    const float* __restrict__ gfw, const float* __restrict__ gfb,
    const float* __restrict__ ghiw, const float* __restrict__ ghib,
    const float* __restrict__ lfw, const float* __restrict__ lfb,
    const float* __restrict__ lhiw, const float* __restrict__ lhib,
    const float* __restrict__ nw, char* __restrict__ ws, int k0)
{
  const int s = blockIdx.x * 256 + threadIdx.x;
  if (s >= SLOTS) return;
  const int k = k0 + blockIdx.y;
  const int* idx = (const int*)(ws + IDX_OFF);
  _Float16* x1 = (_Float16*)(ws + X1_OFF) + ((size_t)blockIdx.y * SLOTS + s) * 32;

  const int ped = idx[s];
  if (ped < 0) {  // guard slot -> zeros (conv zero padding at bin edges)
    v8h z = {};
    *(v8h*)(x1) = z; *(v8h*)(x1 + 8) = z; *(v8h*)(x1 + 16) = z; *(v8h*)(x1 + 24) = z;
    return;
  }
  const int* hb = (const int*)(ws + HDR_OFF);
  const int b = (s >= hb[5]) + (s >= hb[6]) + (s >= hb[7]);
  const float NW[4] = {0.05f, 1.0f, 4.0f, 8.0f};
  const float sc = nw[b] * NW[b];
  const float nz[2] = {noise[k * 2 + 0] * sc, noise[k * 2 + 1] * sc};

  // gather padded inputs: apad[du][c][t+1], zeros at t edges & missing neighbors
  float apad[3][2][10];
#pragma unroll
  for (int du = 0; du < 3; du++) {
    int pn = idx[s + du - 1];
    int pnn = pn >= 0 ? pn : 0;
#pragma unroll
    for (int c = 0; c < 2; c++) {
      apad[du][c][0] = 0.f; apad[du][c][9] = 0.f;
#pragma unroll
      for (int t = 0; t < 8; t++) {
        float val = v[(c * 8 + t) * NPED + pnn] + nz[c];
        apad[du][c][t + 1] = pn >= 0 ? val : 0.f;
      }
    }
  }

  // per-bin stage-1 weights
  float WG[2][2][3][3], GB[2], HI[2][2], HIB[2];
  float LF[2][2][3], LFB[2], LHI[2][2], LHIB[2];
#pragma unroll
  for (int oc = 0; oc < 2; oc++) {
    GB[oc] = gfb[b * 2 + oc]; HIB[oc] = ghib[b * 2 + oc];
    LFB[oc] = lfb[b * 2 + oc]; LHIB[oc] = lhib[b * 2 + oc];
#pragma unroll
    for (int ic = 0; ic < 2; ic++) {
      HI[oc][ic] = ghiw[(b * 2 + oc) * 2 + ic];
      LHI[oc][ic] = lhiw[(b * 2 + oc) * 2 + ic];
#pragma unroll
      for (int kk = 0; kk < 3; kk++) {
        LF[oc][ic][kk] = lfw[((b * 2 + oc) * 2 + ic) * 3 + kk];
#pragma unroll
        for (int kw = 0; kw < 3; kw++)
          WG[oc][ic][kk][kw] = gfw[(((b * 2 + oc) * 2 + ic) * 3 + kk) * 3 + kw];
      }
    }
  }

  __attribute__((aligned(16))) _Float16 oh[32];
#pragma unroll
  for (int oc = 0; oc < 2; oc++) {
#pragma unroll
    for (int t = 0; t < 8; t++) {
      float acc = GB[oc];
#pragma unroll
      for (int ic = 0; ic < 2; ic++)
#pragma unroll
        for (int kh = 0; kh < 3; kh++)
#pragma unroll
          for (int kw = 0; kw < 3; kw++)
            acc += WG[oc][ic][kh][kw] * apad[kw][ic][t + kh];
      float hi = HIB[oc];
#pragma unroll
      for (int ic = 0; ic < 2; ic++) hi += HI[oc][ic] * apad[1][ic][t + 1];
      oh[oc * 8 + t] = (_Float16)(fmaxf(acc, 0.f) + hi);

      float la = LFB[oc];
#pragma unroll
      for (int ic = 0; ic < 2; ic++)
#pragma unroll
        for (int kk = 0; kk < 3; kk++)
          la += LF[oc][ic][kk] * apad[1][ic][t + kk];
      float lh = LHIB[oc];
#pragma unroll
      for (int ic = 0; ic < 2; ic++) lh += LHI[oc][ic] * apad[1][ic][t + 1];
      oh[16 + oc * 8 + t] = (_Float16)(fmaxf(la, 0.f) + lh);
    }
  }
#pragma unroll
  for (int q = 0; q < 4; q++)
    *(v8h*)(x1 + q * 8) = *(v8h*)(oh + q * 8);
}

// ---------------------------------------------------------------------------
// Kernel 3: stage-2 fused GEMM via WMMA, scatter to d_out
// ---------------------------------------------------------------------------
__device__ __forceinline__ v16h load_a16(const _Float16* p) {
  // elts 0..7 <- p[0..7], elts 8..15 <- p[8..15]  (contiguous 16-K run)
  v8h lo = *(const v8h*)p;
  v8h hi = *(const v8h*)(p + 8);
  v16h r;
#pragma unroll
  for (int i = 0; i < 8; i++) { r[i] = lo[i]; r[i + 8] = hi[i]; }
  return r;
}

__global__ __launch_bounds__(32) void k_stage2(
    const char* __restrict__ ws, float* __restrict__ out, int k0)
{
  const int* hdr = (const int*)(ws + HDR_OFF);
  const int t = blockIdx.x;
  if (t >= hdr[13]) return;  // block-uniform exit: EXEC stays all-ones below
  const int k = k0 + blockIdx.y;

  const int b  = (t >= hdr[9]) + (t >= hdr[10]) + (t >= hdr[11]);
  const int lt = t - hdr[8 + b];
  const int s0 = hdr[4 + b] + lt * 16;
  const int nv = hdr[b] - lt * 16;

  const int lane = threadIdx.x;
  const int j = lane & 15, half = lane >> 4;
  const bool valid = j < nv;
  const int sC = valid ? s0 + j : 0;  // invalid columns read the zero guard
  const int sL = valid ? sC - 1 : 0;
  const int sR = valid ? sC + 1 : 0;

  const _Float16* x1 = (const _Float16*)(ws + X1_OFF) + (size_t)blockIdx.y * SLOTS * 32;
  const _Float16* w2 = (const _Float16*)(ws + W2F_OFF) + b * 32 * 64;

  // B operands: feature vec f[64] = [x1g(s-1) | x1g(s) | x1g(s+1) | x1l(s)]
  // Each lane's 16-K run is one contiguous 32B block of a slot.
  const _Float16* pb0 = x1 + (size_t)(half ? sC : sL) * 32;
  const _Float16* pb1 = half ? (x1 + (size_t)sC * 32 + 16) : (x1 + (size_t)sR * 32);
  v16h B0 = *(const v16h*)pb0;
  v16h B1 = *(const v16h*)pb1;

  // A operands: W2f rows m (tile0: 0..15, tile1: 16..31), K windows [0,32),[32,64)
  const int b8 = half * 16;
  const _Float16* ra0 = w2 + (size_t)j * 64;
  const _Float16* ra1 = w2 + (size_t)(16 + j) * 64;
  v16h A00 = load_a16(ra0 + b8);
  v16h A01 = load_a16(ra0 + 32 + b8);
  v16h A10 = load_a16(ra1 + b8);
  v16h A11 = load_a16(ra1 + 32 + b8);

  v8f c0 = {}, c1 = {};
  c0 = __builtin_amdgcn_wmma_f32_16x16x32_f16(false, A00, false, B0, (short)0, c0, false, false);
  c1 = __builtin_amdgcn_wmma_f32_16x16x32_f16(false, A10, false, B0, (short)0, c1, false, false);
  c0 = __builtin_amdgcn_wmma_f32_16x16x32_f16(false, A01, false, B1, (short)0, c0, false, false);
  c1 = __builtin_amdgcn_wmma_f32_16x16x32_f16(false, A11, false, B1, (short)0, c1, false, false);

  if (!valid) return;
  const int ped = ((const int*)(ws + IDX_OFF))[s0 + j];
  const float* bias = (const float*)(ws + BIAS_OFF) + b * 32;

#pragma unroll
  for (int r = 0; r < 8; r++) {          // tile0 rows m = r + 8*half (< 16)
    int m = r + 8 * half;
    int h = m / 12, o = m % 12;
    out[(((size_t)k * 2 + h) * 12 + o) * NPED + ped] = c0[r] + bias[m];
  }
  if (half == 0) {
#pragma unroll
    for (int r = 0; r < 8; r++) {        // tile1 rows 16..23 (24..31 are padding)
      int m = 16 + r;
      int h = m / 12, o = m % 12;
      out[(((size_t)k * 2 + h) * 12 + o) * NPED + ped] = c1[r] + bias[m];
    }
  }
}

// ---------------------------------------------------------------------------
extern "C" void kernel_launch(void* const* d_in, const int* in_sizes, int n_in,
                              void* d_out, int out_size, void* d_ws, size_t ws_size,
                              hipStream_t stream)
{
  const float* v     = (const float*)d_in[0];
  const float* noise = (const float*)d_in[2];
  const float* gfw   = (const float*)d_in[3];
  const float* gfb   = (const float*)d_in[4];
  const float* ghiw  = (const float*)d_in[5];
  const float* ghib  = (const float*)d_in[6];
  const float* ghw   = (const float*)d_in[7];
  const float* ghb   = (const float*)d_in[8];
  const float* gtw   = (const float*)d_in[9];
  const float* gtb   = (const float*)d_in[10];
  const float* lfw   = (const float*)d_in[11];
  const float* lfb   = (const float*)d_in[12];
  const float* lhiw  = (const float*)d_in[13];
  const float* lhib  = (const float*)d_in[14];
  const float* lhw   = (const float*)d_in[15];
  const float* lhb   = (const float*)d_in[16];
  const float* ltw   = (const float*)d_in[17];
  const float* ltb   = (const float*)d_in[18];
  const float* nw    = (const float*)d_in[19];
  const float* gw    = (const float*)d_in[20];
  const float* lw    = (const float*)d_in[21];
  float* out = (float*)d_out;
  char* ws = (char*)d_ws;

  k_setup<<<1, 1024, 0, stream>>>(v, gtw, gtb, ghw, ghb, ltw, ltb, lhw, lhb, gw, lw, ws);

  // k-slicing so the f16 feature scratch fits in whatever ws_size we got
  const size_t perk = (size_t)SLOTS * 32 * 2;  // ~3.2 MB per noise step
  int kpass = 1;
  if (ws_size > (size_t)X1_OFF + perk)
    kpass = (int)((ws_size - (size_t)X1_OFF) / perk);
  if (kpass > KST) kpass = KST;
  if (kpass < 1) kpass = 1;

  for (int ktp = 0; ktp < KST; ktp += kpass) {
    int kc = (KST - ktp < kpass) ? (KST - ktp) : kpass;
    k_stage1<<<dim3((SLOTS + 255) / 256, kc), 256, 0, stream>>>(
        v, noise, gfw, gfb, ghiw, ghib, lfw, lfb, lhiw, lhib, nw, ws, ktp);
    k_stage2<<<dim3(3136, kc), 32, 0, stream>>>(ws, out, ktp);
  }
}
